// ContinuousTreeDecoder_52561809769010
// MI455X (gfx1250) — matrix operand order repaired and checked by reading, not compiled, not run on hardware
//
#include <hip/hip_runtime.h>
#include <stdint.h>

// ---------------------------------------------------------------------------
// ContinuousTreeDecoder on MI455X (gfx1250): single persistent kernel,
// 8 WGs x 512 threads, device-scope barrier between phases, bf16 WMMA GEMMs.
// Round 3: transposed K cache -> fully coalesced attention score reads
// (contiguous 512B wave accesses on the dominant L2 stream).
// ---------------------------------------------------------------------------

typedef __bf16 bf16;
typedef __attribute__((ext_vector_type(16))) __bf16 v16bf;
typedef __attribute__((ext_vector_type(8)))  float  v8f;

#define NWG 8
#define TPB 512
#define NWAVES ((NWG * TPB) / 32)   // 128 waves
#define GTOT (NWG * TPB)            // 4096 threads

// model dims
constexpr int CB = 32, CH = 512, CNH = 8, CFF = 2048, CV = 256;
constexpr int CT = 769, CMT = 768, CLAT = 256, CML = 8, CMPL = 96;
constexpr int TP = 776;             // padded position stride for transposed K

// ---- d_out layout (floats): tokens, states, lengths, mask, segs ----
constexpr size_t TOK_OFF  = 0;
constexpr size_t ST_OFF   = TOK_OFF  + (size_t)CB * CMT;
constexpr size_t LEN_OFF  = ST_OFF   + (size_t)CB * CMT * CH;
constexpr size_t MASK_OFF = LEN_OFF  + (size_t)CB;
constexpr size_t SEG_OFF  = MASK_OFF + (size_t)CB * CMT;
constexpr size_t OUT_TOT  = SEG_OFF  + (size_t)CB * CML;   // 12,632,352 floats

// ---- workspace layout (byte offsets into d_ws) ----
constexpr size_t WS_MEM  = 4096;                              // f32 [32][8][512] memories
constexpr size_t WS_WQT  = WS_MEM  + 4ull * CB * CML * CH;    // bf16 [512][512] (Wq^T)
constexpr size_t WS_WKT  = WS_WQT  + 2ull * CH * CH;
constexpr size_t WS_WVT  = WS_WKT  + 2ull * CH * CH;
constexpr size_t WS_WOT  = WS_WVT  + 2ull * CH * CH;
constexpr size_t WS_W1T  = WS_WOT  + 2ull * CH * CH;          // bf16 [2048][512]
constexpr size_t WS_W2T  = WS_W1T  + 2ull * CH * CFF;         // bf16 [512][2048]
constexpr size_t WS_WOUT = WS_W2T  + 2ull * CFF * CH;         // bf16 [256][512]
constexpr size_t WS_KT   = WS_WOUT + 2ull * CV * CH;          // bf16 [32*8*64][776] (K^T)
constexpr size_t WS_VC   = WS_KT   + 2ull * CB * CNH * 64 * TP;
constexpr size_t WS_HF   = WS_VC   + 2ull * CB * CT * CH;     // f32  h
constexpr size_t WS_HBF  = WS_HF   + 4ull * CB * CH;          // bf16 h (A operand)
constexpr size_t WS_QF   = WS_HBF  + 2ull * CB * CH;          // f32  q
constexpr size_t WS_OBF  = WS_QF   + 4ull * CB * CH;          // bf16 attn out
constexpr size_t WS_RES  = WS_OBF  + 2ull * CB * CH;          // f32  pre-LN buf
constexpr size_t WS_H2F  = WS_RES  + 4ull * CB * CH;          // f32  h2 (post-LN1)
constexpr size_t WS_ABF  = WS_H2F  + 4ull * CB * CH;          // bf16 [32][2048] ffn act

struct Ctx {
  const float *zf, *wlat, *blat, *emb, *wq, *wk, *wv, *wo, *w1, *w2, *wout;
  const unsigned char *act;
  float *out;
  float *mem, *hf, *qf, *resid, *h2f;
  bf16  *wqt, *wkt, *wvt, *wot, *w1t, *w2t, *woutt, *kt, *vc;
  bf16  *hbf, *obf, *abf;
  int   *cur_leaf, *tis, *fin, *tpos, *nxt, *wflag, *pslot;
  unsigned *bar;
  float (*s_sc)[772];
  float (*s_q)[64];
  bf16  (*s_h2bf)[CH];
  bf16  (*s_outbf)[CH];
  int gtid, gw, lane, wib;
};

__device__ __forceinline__ float bfu(unsigned u) {   // bf16 bits (low 16) -> float
  u <<= 16;
  return __builtin_bit_cast(float, u);
}
__device__ __forceinline__ float wave_max(float v) {
  for (int o = 16; o; o >>= 1) v = fmaxf(v, __shfl_xor(v, o, 32));
  return v;
}
__device__ __forceinline__ float wave_sum(float v) {
  for (int o = 16; o; o >>= 1) v += __shfl_xor(v, o, 32);
  return v;
}

__device__ __forceinline__ void gsync(const Ctx& c, unsigned& gen) {
  __threadfence();
  __syncthreads();
  if (threadIdx.x == 0) {
    ++gen;
    __hip_atomic_fetch_add(c.bar, 1u, __ATOMIC_ACQ_REL, __HIP_MEMORY_SCOPE_AGENT);
    while (__hip_atomic_load(c.bar, __ATOMIC_ACQUIRE, __HIP_MEMORY_SCOPE_AGENT) <
           gen * (unsigned)NWG)
      __builtin_amdgcn_s_sleep(1);
  }
  __syncthreads();
  __threadfence();
}

struct Frag32 { uint4 lo, hi; };

// 16x16 f32 tile of A(row-major bf16, M x K) @ Bt(row-major [N][K] bf16)^T
// A may point to LDS (address space inferred after inlining) or global.
__device__ __forceinline__ v8f wmma_mm(const bf16* A, int lda, const bf16* Bt, int ldb,
                                       int m0, int n0, int K, int lane) {
  v8f acc = {};
  const int kh = lane >> 4;
  const bf16* pa0 = A  + (size_t)(m0 + (lane & 15)) * lda + kh * 8;
  const bf16* pb0 = Bt + (size_t)(n0 + (lane & 15)) * ldb + kh * 8;
  for (int k0 = 0; k0 < K; k0 += 32) {
    Frag32 fa, fb;
    fa.lo = *(const uint4*)(pa0 + k0);
    fa.hi = *(const uint4*)(pa0 + k0 + 16);
    fb.lo = *(const uint4*)(pb0 + k0);
    fb.hi = *(const uint4*)(pb0 + k0 + 16);
    v16bf av = __builtin_bit_cast(v16bf, fa);
    v16bf bv = __builtin_bit_cast(v16bf, fb);
    acc = __builtin_amdgcn_wmma_f32_16x16x32_bf16(false, av, false, bv,
                                                  (short)0, acc, false, false);
  }
  return acc;
}

// ---------------- phases ----------------

__device__ __forceinline__ void phase_init(Ctx& c) {
  float4* o4 = (float4*)c.out;
  const size_t n4 = OUT_TOT / 4;
  for (size_t i = c.gtid; i < n4; i += GTOT) o4[i] = make_float4(0.f, 0.f, 0.f, 0.f);
  for (int idx = c.gtid; idx < CB * CML * CH; idx += GTOT) {   // memories
    int bl = idx >> 9, hh = idx & 511;
    const float* z = c.zf + (size_t)bl * CLAT;
    float s = c.blat[hh];
    for (int d = 0; d < CLAT; ++d) s += z[d] * c.wlat[(size_t)d * CH + hh];
    c.mem[idx] = s;
  }
  for (int idx = c.gtid; idx < CH * CH; idx += GTOT) {         // Wq/k/v/o^T -> bf16
    int n = idx >> 9, k = idx & 511;
    size_t src = (size_t)k * CH + n;
    c.wqt[idx] = (bf16)c.wq[src];
    c.wkt[idx] = (bf16)c.wk[src];
    c.wvt[idx] = (bf16)c.wv[src];
    c.wot[idx] = (bf16)c.wo[src];
  }
  for (int idx = c.gtid; idx < CFF * CH; idx += GTOT) {        // W1t [2048][512]
    int n = idx >> 9, k = idx & 511;
    c.w1t[idx] = (bf16)c.w1[(size_t)k * CFF + n];
  }
  for (int idx = c.gtid; idx < CH * CFF; idx += GTOT) {        // W2t [512][2048]
    int n = idx >> 11, k = idx & 2047;
    c.w2t[idx] = (bf16)c.w2[(size_t)k * CH + n];
  }
  for (int idx = c.gtid; idx < CV * CH; idx += GTOT) {         // Woutt [256][512]
    int n = idx >> 9, k = idx & 511;
    c.woutt[idx] = (bf16)c.wout[(size_t)k * CV + n];
  }
}

__device__ __forceinline__ void phase_prime(Ctx& c) {     // h = embed[BOS] + mem(leaf 0)
  for (int idx = c.gtid; idx < CB * CH; idx += GTOT) {
    int b = idx >> 9, hh = idx & 511;
    float m = c.act[b * CML] ? c.mem[(size_t)(b * CML) * CH + hh] : 0.f;
    float hv = c.emb[(size_t)1 * CH + hh] + m;
    c.hf[idx] = hv;
    c.hbf[idx] = (bf16)hv;
  }
}

__device__ __forceinline__ void phase_qkv(Ctx& c, int pos) {
  for (int tile = c.gw; tile < 192; tile += NWAVES) {
    int mat = tile / 64, tl = tile % 64;
    int m0 = (tl & 1) * 16, n0 = (tl >> 1) * 16;
    const bf16* Bt = (mat == 0) ? c.wqt : (mat == 1) ? c.wkt : c.wvt;
    v8f acc = wmma_mm(c.hbf, CH, Bt, CH, m0, n0, CH, c.lane);
    int n = n0 + (c.lane & 15);
    int mb = m0 + ((c.lane >> 4) << 3);
    if (mat == 0) {
      for (int r = 0; r < 8; ++r) c.qf[(size_t)(mb + r) * CH + n] = acc[r];
    } else if (mat == 1) {
      int hn = n >> 6, d = n & 63;       // K cache transposed: [(b,h,d)][pos]
      for (int r = 0; r < 8; ++r)
        c.kt[((size_t)((mb + r) * CNH + hn) * 64 + d) * TP + pos] = (bf16)acc[r];
    } else {
      for (int r = 0; r < 8; ++r)
        c.vc[((size_t)(mb + r) * CT + pos) * CH + n] = (bf16)acc[r];
    }
  }
}

__device__ __forceinline__ void phase_attn(Ctx& c, int pos) {
  const int L = pos + 1;
  for (int job = c.gw; job < CB * CNH; job += NWAVES) {
    int b = job >> 3, hn = job & 7;
    // q into per-wave LDS (broadcast reads inside the dim loop)
    c.s_q[c.wib][c.lane]      = c.qf[(size_t)b * CH + hn * 64 + c.lane];
    c.s_q[c.wib][c.lane + 32] = c.qf[(size_t)b * CH + hn * 64 + 32 + c.lane];
    // pass1: each lane owns 8 consecutive positions -> contiguous 512B wave loads
    const bf16* kTb = c.kt + (size_t)(b * CNH + hn) * 64 * TP;
    float mx = -3.4e38f;
    for (int base = 0; base < L; base += 256) {
      const bf16* kp = kTb + base + c.lane * 8;
      float s0 = 0.f, s1 = 0.f, s2 = 0.f, s3 = 0.f,
            s4 = 0.f, s5 = 0.f, s6 = 0.f, s7 = 0.f;
      for (int d = 0; d < 64; ++d) {
        uint4 u = *(const uint4*)(kp + (size_t)d * TP);
        float qd = c.s_q[c.wib][d];
        s0 += qd * bfu(u.x & 0xffffu); s1 += qd * bfu(u.x >> 16);
        s2 += qd * bfu(u.y & 0xffffu); s3 += qd * bfu(u.y >> 16);
        s4 += qd * bfu(u.z & 0xffffu); s5 += qd * bfu(u.z >> 16);
        s6 += qd * bfu(u.w & 0xffffu); s7 += qd * bfu(u.w >> 16);
      }
      float sv[8] = {s0, s1, s2, s3, s4, s5, s6, s7};
      int p0 = base + c.lane * 8;
#pragma unroll
      for (int j = 0; j < 8; ++j) {
        int p = p0 + j;
        if (p < L) {
          float v = sv[j] * 0.125f;     // 1/sqrt(64)
          c.s_sc[c.wib][p] = v;
          mx = fmaxf(mx, v);
        }
      }
    }
    mx = wave_max(mx);
    // pass2: exp + sum
    float ls = 0.f;
    for (int p = c.lane; p < L; p += 32) {
      float e = __expf(c.s_sc[c.wib][p] - mx);
      c.s_sc[c.wib][p] = e;
      ls += e;
    }
    ls = wave_sum(ls);
    float inv = 1.f / ls;
    // pass3: o = attn @ V (lanes own 2 dims; 2 independent acc pairs)
    const unsigned* vb = (const unsigned*)(c.vc + (size_t)b * CT * CH + hn * 64) + c.lane;
    float o0 = 0.f, o1 = 0.f, e0 = 0.f, e1 = 0.f;
    int p = 0;
    for (; p + 1 < L; p += 2) {
      float a0 = c.s_sc[c.wib][p], a1 = c.s_sc[c.wib][p + 1];
      unsigned u0 = vb[(size_t)p * (CH / 2)], u1 = vb[(size_t)(p + 1) * (CH / 2)];
      o0 += a0 * bfu(u0 & 0xffffu); o1 += a0 * bfu(u0 >> 16);
      e0 += a1 * bfu(u1 & 0xffffu); e1 += a1 * bfu(u1 >> 16);
    }
    if (p < L) {
      float a0 = c.s_sc[c.wib][p];
      unsigned u0 = vb[(size_t)p * (CH / 2)];
      o0 += a0 * bfu(u0 & 0xffffu); o1 += a0 * bfu(u0 >> 16);
    }
    o0 += e0; o1 += e1;
    int d0 = hn * 64 + c.lane * 2;
    c.obf[(size_t)b * CH + d0]     = (bf16)(o0 * inv);
    c.obf[(size_t)b * CH + d0 + 1] = (bf16)(o1 * inv);
  }
}

__device__ __forceinline__ void phase_wo(Ctx& c) {
  for (int tile = c.gw; tile < 64; tile += NWAVES) {
    int m0 = (tile & 1) * 16, n0 = (tile >> 1) * 16;
    v8f acc = wmma_mm(c.obf, CH, c.wot, CH, m0, n0, CH, c.lane);
    int n = n0 + (c.lane & 15);
    int mb = m0 + ((c.lane >> 4) << 3);
    for (int r = 0; r < 8; ++r) {
      size_t idx = (size_t)(mb + r) * CH + n;
      c.resid[idx] = c.hf[idx] + acc[r];
    }
  }
}

// FFN1 with folded LN1: every block recomputes LN redundantly into its LDS,
// block 0 also publishes h2 (f32) globally for the FFN2 residual.
__device__ __forceinline__ void phase_ffn1(Ctx& c) {
  for (int b = c.wib; b < CB; b += TPB / 32) {
    float sum = 0.f, sq = 0.f;
    for (int i = c.lane; i < CH; i += 32) {
      float v = c.resid[(size_t)b * CH + i];
      sum += v; sq += v * v;
    }
    sum = wave_sum(sum); sq = wave_sum(sq);
    float m = sum * (1.f / CH);
    float var = sq * (1.f / CH) - m * m;
    float r = rsqrtf(var + 1e-5f);
    for (int i = c.lane; i < CH; i += 32) {
      float y = (c.resid[(size_t)b * CH + i] - m) * r;
      c.s_h2bf[b][i] = (bf16)y;
      if (blockIdx.x == 0) c.h2f[(size_t)b * CH + i] = y;
    }
  }
  __syncthreads();
  for (int tile = c.gw; tile < 256; tile += NWAVES) {
    int m0 = (tile & 1) * 16, n0 = (tile >> 1) * 16;
    v8f acc = wmma_mm(&c.s_h2bf[0][0], CH, c.w1t, CH, m0, n0, CH, c.lane);
    int n = n0 + (c.lane & 15);
    int mb = m0 + ((c.lane >> 4) << 3);
    for (int r = 0; r < 8; ++r)
      c.abf[(size_t)(mb + r) * CFF + n] = (bf16)fmaxf(acc[r], 0.f);
  }
}

__device__ __forceinline__ void phase_ffn2(Ctx& c) {
  for (int tile = c.gw; tile < 64; tile += NWAVES) {
    int m0 = (tile & 1) * 16, n0 = (tile >> 1) * 16;
    v8f acc = wmma_mm(c.abf, CFF, c.w2t, CFF, m0, n0, CFF, c.lane);
    int n = n0 + (c.lane & 15);
    int mb = m0 + ((c.lane >> 4) << 3);
    for (int r = 0; r < 8; ++r) {
      size_t idx = (size_t)(mb + r) * CH + n;
      c.resid[idx] = c.h2f[idx] + acc[r];
    }
  }
}

// Fused tail on block 0 only: LN2 (rows in regs) -> logits (LDS A, logits into
// score-LDS) -> argmax/bookkeeping -> states copy -> next-step h prep.
__device__ __forceinline__ void phase_tail(Ctx& c) {
  if (blockIdx.x != 0) return;
  float (*slog)[CV] = (float (*)[CV]) & c.s_sc[0][0];   // reuse score LDS
  float ya[16], yb[16];
  const int b0 = c.wib, b1 = c.wib + 16;
  {
    float sum = 0.f, sq = 0.f;
    for (int i = c.lane; i < CH; i += 32) {
      float v = c.resid[(size_t)b0 * CH + i];
      sum += v; sq += v * v;
    }
    sum = wave_sum(sum); sq = wave_sum(sq);
    float m = sum * (1.f / CH), var = sq * (1.f / CH) - m * m, r = rsqrtf(var + 1e-5f);
#pragma unroll
    for (int j = 0; j < 16; ++j) {
      int i = c.lane + j * 32;
      float y = (c.resid[(size_t)b0 * CH + i] - m) * r;
      ya[j] = y;
      c.s_outbf[b0][i] = (bf16)y;
    }
  }
  {
    float sum = 0.f, sq = 0.f;
    for (int i = c.lane; i < CH; i += 32) {
      float v = c.resid[(size_t)b1 * CH + i];
      sum += v; sq += v * v;
    }
    sum = wave_sum(sum); sq = wave_sum(sq);
    float m = sum * (1.f / CH), var = sq * (1.f / CH) - m * m, r = rsqrtf(var + 1e-5f);
#pragma unroll
    for (int j = 0; j < 16; ++j) {
      int i = c.lane + j * 32;
      float y = (c.resid[(size_t)b1 * CH + i] - m) * r;
      yb[j] = y;
      c.s_outbf[b1][i] = (bf16)y;
    }
  }
  __syncthreads();
  for (int tile = c.wib; tile < 32; tile += TPB / 32) {   // logits GEMM (LDS A)
    int m0 = (tile & 1) * 16, n0 = (tile >> 1) * 16;
    v8f acc = wmma_mm(&c.s_outbf[0][0], CH, c.woutt, CH, m0, n0, CH, c.lane);
    int n = n0 + (c.lane & 15);
    int mb = m0 + ((c.lane >> 4) << 3);
    for (int r = 0; r < 8; ++r) slog[mb + r][n] = acc[r];
  }
  __syncthreads();
  if (c.wib == 0) {                                       // bookkeeping, lane == batch
    int b = c.lane;
    float best = slog[b][0]; int bi = 0;
    for (int j = 1; j < CV; ++j) { float v = slog[b][j]; if (v > best) { best = v; bi = j; } }
    int fin = c.fin[b];
    int w = !fin;
    int tp = c.tpos[b];
    int p = tp > (CMT - 1) ? (CMT - 1) : tp;
    if (w) {
      c.out[TOK_OFF  + (size_t)b * CMT + p] = (float)bi;
      c.out[MASK_OFF + (size_t)b * CMT + p] = 1.0f;
    }
    tp += w;
    int ts = c.tis[b] + w;
    int sw = w && (bi == 2 /*EOS*/ || ts >= CMPL);
    int cl2 = c.cur_leaf[b] + sw;
    ts = sw ? 0 : ts;
    int sl = cl2 > (CML - 1) ? (CML - 1) : cl2;
    if (sw && cl2 < CML) c.out[SEG_OFF + (size_t)b * CML + sl] = (float)tp;
    int clm = cl2 > (CML - 1) ? (CML - 1) : cl2;
    int finNew = fin | (sw && (cl2 >= CML || !c.act[b * CML + clm]));
    c.cur_leaf[b] = cl2; c.tis[b] = ts; c.fin[b] = finNew; c.tpos[b] = tp;
    c.nxt[b] = bi; c.wflag[b] = w; c.pslot[b] = p;
  }
  __syncthreads();
  // states copy straight from LN2 registers
  if (c.wflag[b0]) {
    float* dst = c.out + ST_OFF + ((size_t)b0 * CMT + c.pslot[b0]) * CH;
#pragma unroll
    for (int j = 0; j < 16; ++j) dst[c.lane + j * 32] = ya[j];
  }
  if (c.wflag[b1]) {
    float* dst = c.out + ST_OFF + ((size_t)b1 * CMT + c.pslot[b1]) * CH;
#pragma unroll
    for (int j = 0; j < 16; ++j) dst[c.lane + j * 32] = yb[j];
  }
  // next-step h = embed[nxt] + mem(cur_leaf)
  for (int idx = threadIdx.x; idx < CB * CH; idx += TPB) {
    int b = idx >> 9, hh = idx & 511;
    int cl = c.cur_leaf[b];
    int li = cl > (CML - 1) ? (CML - 1) : cl;
    float m = (cl < CML && c.act[b * CML + li]) ? c.mem[(size_t)(b * CML + li) * CH + hh] : 0.f;
    float hv = c.emb[(size_t)c.nxt[b] * CH + hh] + m;
    c.hf[idx] = hv;
    c.hbf[idx] = (bf16)hv;
  }
}

__device__ __forceinline__ void dec_step(Ctx& c, int pos, unsigned& gen) {
  phase_qkv(c, pos);    gsync(c, gen);
  phase_attn(c, pos);   gsync(c, gen);
  phase_wo(c);          gsync(c, gen);
  phase_ffn1(c);        gsync(c, gen);   // includes folded LN1
  phase_ffn2(c);        gsync(c, gen);   // leaves pre-LN 'out' in resid
}

// ---------------- persistent kernel ----------------

__global__ void __launch_bounds__(TPB, 1)
ctd_persistent(const float* zf, const unsigned char* act, const float* wlat,
               const float* blat, const float* emb, const float* wq, const float* wk,
               const float* wv, const float* wo, const float* w1, const float* w2,
               const float* wout, float* out, char* ws) {
  __shared__ float s_sc[TPB / 32][772];     //  49,408 B (reused for logits in tail)
  __shared__ float s_q[TPB / 32][64];       //   4,096 B (per-wave q vector)
  __shared__ bf16  s_h2bf[CB][CH];          //  32,768 B
  __shared__ bf16  s_outbf[CB][CH];         //  32,768 B

  Ctx c;
  c.zf = zf; c.act = act; c.wlat = wlat; c.blat = blat; c.emb = emb;
  c.wq = wq; c.wk = wk; c.wv = wv; c.wo = wo; c.w1 = w1; c.w2 = w2; c.wout = wout;
  c.out = out;
  c.bar      = (unsigned*)(ws);
  c.cur_leaf = (int*)(ws + 128);
  c.tis      = (int*)(ws + 256);
  c.fin      = (int*)(ws + 384);
  c.tpos     = (int*)(ws + 512);
  c.nxt      = (int*)(ws + 640);
  c.wflag    = (int*)(ws + 768);
  c.pslot    = (int*)(ws + 896);
  c.mem   = (float*)(ws + WS_MEM);
  c.wqt   = (bf16*)(ws + WS_WQT);  c.wkt = (bf16*)(ws + WS_WKT);
  c.wvt   = (bf16*)(ws + WS_WVT);  c.wot = (bf16*)(ws + WS_WOT);
  c.w1t   = (bf16*)(ws + WS_W1T);  c.w2t = (bf16*)(ws + WS_W2T);
  c.woutt = (bf16*)(ws + WS_WOUT);
  c.kt    = (bf16*)(ws + WS_KT);   c.vc  = (bf16*)(ws + WS_VC);
  c.hf    = (float*)(ws + WS_HF);  c.hbf = (bf16*)(ws + WS_HBF);
  c.qf    = (float*)(ws + WS_QF);  c.obf = (bf16*)(ws + WS_OBF);
  c.resid = (float*)(ws + WS_RES); c.h2f = (float*)(ws + WS_H2F);
  c.abf   = (bf16*)(ws + WS_ABF);
  c.s_sc    = s_sc;
  c.s_q     = s_q;
  c.s_h2bf  = s_h2bf;
  c.s_outbf = s_outbf;
  c.gtid = blockIdx.x * TPB + threadIdx.x;
  c.gw   = c.gtid >> 5;
  c.lane = threadIdx.x & 31;
  c.wib  = threadIdx.x >> 5;

  unsigned gen = 0;

  phase_init(c);   gsync(c, gen);
  phase_prime(c);  gsync(c, gen);

  dec_step(c, 0, gen);                 // BOS priming step -> pre-LN out0 in resid

#pragma unroll 1
  for (int t = 0; t < CMT; ++t) {
    phase_tail(c);                     // LN2 + logits + argmax + book + prep (block 0)
    gsync(c, gen);
    if (t < CMT - 1) dec_step(c, t + 1, gen);   // final step's decode is dead code
  }

  if (blockIdx.x == 0 && c.wib == 0)
    out[LEN_OFF + c.lane] = (float)c.tpos[c.lane];     // lengths
}

// ---------------- host entry ----------------

extern "C" void kernel_launch(void* const* d_in, const int* in_sizes, int n_in,
                              void* d_out, int out_size, void* d_ws, size_t ws_size,
                              hipStream_t stream) {
  (void)in_sizes; (void)n_in; (void)out_size; (void)ws_size;
  // reset the device-scope barrier counter + bookkeeping ints (graph-capture safe)
  hipMemsetAsync(d_ws, 0, 4096, stream);
  ctd_persistent<<<dim3(NWG), dim3(TPB), 0, stream>>>(
      (const float*)d_in[0],          // leaf_z
      (const unsigned char*)d_in[1],  // leaf_active (bool, 1 byte)
      (const float*)d_in[2],          // W_lat
      (const float*)d_in[3],          // b_lat
      (const float*)d_in[4],          // embed
      (const float*)d_in[5],          // Wq
      (const float*)d_in[6],          // Wk
      (const float*)d_in[7],          // Wv
      (const float*)d_in[8],          // Wo
      (const float*)d_in[9],          // W1
      (const float*)d_in[10],         // W2
      (const float*)d_in[11],         // W_out
      (float*)d_out, (char*)d_ws);
}
